// GCN_68453188764220
// MI455X (gfx1250) — matrix operand order repaired and checked by reading, not compiled, not run on hardware
//
#include <hip/hip_runtime.h>
#include <hip/hip_bf16.h>
#include <math.h>

// ---------------------------------------------------------------------------
// CDNA5 (gfx1250) GCN forward pass.
//   GEMMs: v_wmma_f32_16x16x32_bf16. 256-thread blocks = 8 waves, one 16x64
//          output tile per wave. bf16 operands staged once (activations +
//          transposed weights zero-padded to 64-col multiples), so the inner
//          loop is branch-free: 2+8 global_load_b128 + 4 WMMA per 32-K step.
//   Edge aggregation: float4 gather + global_atomic_add_f32 scatter
//          (xw <= 102MB is resident in the 192MB L2).
// ---------------------------------------------------------------------------

typedef __bf16 bf16;
typedef bf16  v16bf __attribute__((ext_vector_type(16)));
typedef float v8f   __attribute__((ext_vector_type(8)));

#define N_NODES 50000
#define N_EDGES 800000
#define GEMM_WAVES 8   // waves (M-tiles) per block

__device__ __forceinline__ void atomic_add_f32(float* p, float v) {
  __hip_atomic_fetch_add(p, v, __ATOMIC_RELAXED, __HIP_MEMORY_SCOPE_AGENT);
}

// ---------------------------------------------------------------------------
__global__ void k_fill(float* __restrict__ p, float v, size_t n) {
  size_t i = (size_t)blockIdx.x * blockDim.x + threadIdx.x;
  if (i < n) p[i] = v;
}

// f32 -> bf16 straight conversion (activations)
__global__ void k_cvt_bf16(const float* __restrict__ in, bf16* __restrict__ out, unsigned n) {
  unsigned i = blockIdx.x * blockDim.x + threadIdx.x;
  if (i < n) out[i] = (bf16)in[i];
}

// W[K,N] f32 row-major -> Wt[Npad,K] bf16 row-major (transpose + convert),
// rows n in [N, Npad) zero-filled so GEMM B-loads never need a guard.
__global__ void k_transpose_cvt_pad(const float* __restrict__ W, bf16* __restrict__ Wt,
                                    unsigned K, unsigned N, unsigned Npad) {
  unsigned idx = blockIdx.x * blockDim.x + threadIdx.x;
  if (idx >= Npad * K) return;
  unsigned n = idx / K;
  unsigned k = idx - n * K;
  Wt[idx] = (n < N) ? (bf16)W[k * N + n] : (bf16)0.0f;
}

// deg[dst] += ew  (one thread per edge)
__global__ void k_deg_scatter(const long long* __restrict__ dst,
                              const float* __restrict__ ew,
                              float* __restrict__ deg, int E) {
  int e = blockIdx.x * blockDim.x + threadIdx.x;
  if (e < E) atomic_add_f32(deg + dst[e], ew[e]);
}

__global__ void k_rsqrt(const float* __restrict__ deg, float* __restrict__ dinv, int n) {
  int i = blockIdx.x * blockDim.x + threadIdx.x;
  if (i < n) dinv[i] = rsqrtf(deg[i]);
}

// ---------------------------------------------------------------------------
// WMMA GEMM: out[M,N] = act(A[M,K](bf16) @ Wt[Npad,K](bf16)^T + bias)
//   8 waves per block, one 16(M) x 64(N) tile per wave.
//   M % 16 == 0, K % 32 == 0, Wt padded to Npad = ceil(N/64)*64 rows.
//   A fragment  = two 16B contiguous runs per lane (ISA 7.12.2 16-bit A layout)
//   B fragment  = two 16B contiguous runs per lane (Wt row = column of W)
__global__ void k_wmma_gemm(const bf16* __restrict__ A, const bf16* __restrict__ Wt,
                            const float* __restrict__ bias,
                            float* __restrict__ outf, bf16* __restrict__ outb,
                            int M, int N, int K, int do_relu) {
  const int wave  = threadIdx.x >> 5;
  const int lane  = threadIdx.x & 31;
  const int half  = lane >> 4;      // 0: lanes 0-15, 1: lanes 16-31
  const int l16   = lane & 15;
  const int mtile = blockIdx.x * GEMM_WAVES + wave;
  if (mtile >= (M >> 4)) return;    // wave-uniform: EXEC stays all-ones below
  const int m0 = mtile << 4;
  const int n0 = blockIdx.y << 6;

  v8f acc[4] = {};
  const unsigned abase = (unsigned)(m0 + l16) * (unsigned)K;

  for (int k0 = 0; k0 < K; k0 += 32) {
    // ---- A fragment: lanes 0-15 hold K = k0+0..7 / k0+16..23 (+8 for 16-31)
    v16bf afrag;
    {
      unsigned a0 = abase + (unsigned)(k0 + (half << 3));
      ((uint4*)&afrag)[0] = *(const uint4*)(A + a0);        // K offsets +0..7
      ((uint4*)&afrag)[1] = *(const uint4*)(A + a0 + 16);   // K offsets +16..23
    }

    // ---- 4 B fragments (always in-bounds: Wt is zero-padded), reuse afrag
#pragma unroll
    for (int t = 0; t < 4; ++t) {
      unsigned ncol = (unsigned)(n0 + (t << 4) + l16);
      unsigned b0   = ncol * (unsigned)K + (unsigned)(k0 + (half << 3));
      v16bf bfrag;
      ((uint4*)&bfrag)[0] = *(const uint4*)(Wt + b0);       // K +0..7
      ((uint4*)&bfrag)[1] = *(const uint4*)(Wt + b0 + 16);  // K +16..23
      acc[t] = __builtin_amdgcn_wmma_f32_16x16x32_bf16(
          false, afrag, false, bfrag, (short)0, acc[t], false, false);
    }
  }

  // ---- store: C/D layout -> lane col = l16, VGPR v -> row m0 + v + 8*half
#pragma unroll
  for (int t = 0; t < 4; ++t) {
    int ncol = n0 + (t << 4) + l16;
    if (ncol >= N) continue;        // only head GEMMs (N=40/32) take this
    float bv = bias ? bias[ncol] : 0.0f;
#pragma unroll
    for (int v = 0; v < 8; ++v) {
      unsigned off = (unsigned)(m0 + v + (half << 3)) * (unsigned)N + (unsigned)ncol;
      float val = acc[t][v] + bv;
      if (do_relu) val = fmaxf(val, 0.0f);
      if (outb) outb[off] = (bf16)val;
      else      outf[off] = val;
    }
  }
}

// ---------------------------------------------------------------------------
// Edge scatter: agg[dst,:] += (dinv[src]*ew*dinv[dst]) * xw[src,:]
//   One block per edge, blockDim = C/4, float4 gather + f32 atomics (L2-resident).
__global__ void k_edge_scatter(const long long* __restrict__ src,
                               const long long* __restrict__ dst,
                               const float* __restrict__ ew,
                               const float* __restrict__ dinv,
                               const float* __restrict__ xw,
                               float* __restrict__ agg, int E, int C) {
  int e = blockIdx.x;
  if (e >= E) return;
  long long s = src[e];
  long long d = dst[e];
  float norm = dinv[s] * ew[e] * dinv[d];
  const float4* xs = (const float4*)(xw + (size_t)s * C);
  float* ad = agg + (size_t)d * C;
  int C4 = C >> 2;
  for (int c = threadIdx.x; c < C4; c += blockDim.x) {
    float4 v = xs[c];
    int b = c << 2;
    atomic_add_f32(ad + b,     norm * v.x);
    atomic_add_f32(ad + b + 1, norm * v.y);
    atomic_add_f32(ad + b + 2, norm * v.z);
    atomic_add_f32(ad + b + 3, norm * v.w);
  }
}

// h = relu(agg + dinv^2 * xw + bias)  emitted directly as bf16 for the next GEMM.
// C = 1 << logC (power of two for all conv layers).
__global__ void k_selfloop_bias_relu(const float* __restrict__ agg,
                                     const float* __restrict__ xw,
                                     const float* __restrict__ dinv,
                                     const float* __restrict__ bias,
                                     bf16* __restrict__ hout, int n, int logC) {
  unsigned idx = blockIdx.x * blockDim.x + threadIdx.x;
  unsigned total = (unsigned)n << logC;
  if (idx >= total) return;
  unsigned node = idx >> logC;
  unsigned c    = idx & ((1u << logC) - 1u);
  float di = dinv[node];
  float v  = agg[idx] + di * di * xw[idx] + bias[c];
  hout[idx] = (bf16)fmaxf(v, 0.0f);
}

// Row softmax over C classes (C = 40), one thread per node.
__global__ void k_softmax(const float* __restrict__ z, float* __restrict__ out,
                          int n, int C) {
  int i = blockIdx.x * blockDim.x + threadIdx.x;
  if (i >= n) return;
  const float* zi = z + (size_t)i * C;
  float m = -INFINITY;
  for (int c = 0; c < C; ++c) m = fmaxf(m, zi[c]);
  float s = 0.0f;
  for (int c = 0; c < C; ++c) s += __expf(zi[c] - m);
  float inv = 1.0f / s;
  float* oi = out + (size_t)i * C;
  for (int c = 0; c < C; ++c) oi[c] = __expf(zi[c] - m) * inv;
}

// ---------------------------------------------------------------------------
static inline void launch_gemm(const bf16* A, const bf16* Wt, const float* b,
                               float* outf, bf16* outb, int M, int N, int K,
                               int relu, hipStream_t s) {
  int mtiles = M / 16;
  dim3 grid((mtiles + GEMM_WAVES - 1) / GEMM_WAVES, (N + 63) / 64);
  k_wmma_gemm<<<grid, 32 * GEMM_WAVES, 0, s>>>(A, Wt, b, outf, outb, M, N, K, relu);
}

static inline void launch_fill(float* p, float v, size_t n, hipStream_t s) {
  k_fill<<<(unsigned)((n + 255) / 256), 256, 0, s>>>(p, v, n);
}

static inline void launch_tcvt(const float* W, bf16* Wt, unsigned K, unsigned N,
                               unsigned Npad, hipStream_t s) {
  unsigned tot = Npad * K;
  k_transpose_cvt_pad<<<(tot + 255) / 256, 256, 0, s>>>(W, Wt, K, N, Npad);
}

extern "C" void kernel_launch(void* const* d_in, const int* in_sizes, int n_in,
                              void* d_out, int out_size, void* d_ws, size_t ws_size,
                              hipStream_t stream) {
  (void)in_sizes; (void)n_in; (void)out_size; (void)ws_size;

  const float*     x    = (const float*)d_in[0];
  const long long* ei   = (const long long*)d_in[1];   // [2, E] int64
  const float*     ew   = (const float*)d_in[2];
  const float*     W1   = (const float*)d_in[3];
  const float*     b1   = (const float*)d_in[4];
  const float*     W2   = (const float*)d_in[5];
  const float*     b2   = (const float*)d_in[6];
  const float*     W3   = (const float*)d_in[7];
  const float*     b3   = (const float*)d_in[8];
  const float*     L1w  = (const float*)d_in[9];
  const float*     L1b  = (const float*)d_in[10];
  const float*     L2w  = (const float*)d_in[11];
  const float*     L2b  = (const float*)d_in[12];
  const float*     L3w  = (const float*)d_in[13];
  const float*     L3b  = (const float*)d_in[14];
  float*           outp = (float*)d_out;

  const long long* src = ei;
  const long long* dst = ei + N_EDGES;

  // ---- workspace layout (BUF = 50000*512 elements) -------------------------
  const size_t BUF = (size_t)N_NODES * 512;
  float* xw   = (float*)d_ws;            // [BUF] f32   (post-GEMM features / logits)
  float* agg  = xw + BUF;                // [BUF] f32   (scatter accumulator)
  bf16*  hA   = (bf16*)(agg + BUF);      // [BUF] bf16  (activation ping)
  bf16*  hB   = hA + BUF;                // [BUF] bf16  (activation pong)
  float* deg  = (float*)(hB + BUF);
  float* dinv = deg + N_NODES;
  bf16*  wts  = (bf16*)(dinv + N_NODES); // transposed, padded bf16 weights
  bf16* W1t = wts;            // [512][128]
  bf16* W2t = W1t + 512*128;  // [256][512]
  bf16* W3t = W2t + 256*512;  // [128][256]
  bf16* L1t = W3t + 128*256;  // [ 64][128]
  bf16* L2t = L1t + 64*128;   // [ 64][ 64]  (N=32 padded to 64)
  bf16* L3t = L2t + 64*64;    // [ 64][ 32]  (N=40 padded to 64)

  // ---- preprocess: weights -> bf16 [Npad][K]; x -> bf16 --------------------
  launch_tcvt(W1, W1t, 128, 512, 512, stream);
  launch_tcvt(W2, W2t, 512, 256, 256, stream);
  launch_tcvt(W3, W3t, 256, 128, 128, stream);
  launch_tcvt(L1w, L1t, 128, 64, 64, stream);
  launch_tcvt(L2w, L2t, 64, 32, 64, stream);
  launch_tcvt(L3w, L3t, 32, 40, 64, stream);
  {
    unsigned n = (unsigned)N_NODES * 128;
    k_cvt_bf16<<<(n + 255) / 256, 256, 0, stream>>>(x, hA, n);
  }

  // ---- degrees: deg = 1 (self-loop) + scatter(ew, dst); dinv = rsqrt(deg) --
  launch_fill(deg, 1.0f, N_NODES, stream);
  k_deg_scatter<<<(N_EDGES + 255) / 256, 256, 0, stream>>>(dst, ew, deg, N_EDGES);
  k_rsqrt<<<(N_NODES + 255) / 256, 256, 0, stream>>>(deg, dinv, N_NODES);

  // ---- GCN layer 1: 128 -> 512 --------------------------------------------
  launch_gemm(hA, W1t, nullptr, xw, nullptr, N_NODES, 512, 128, 0, stream);
  launch_fill(agg, 0.0f, (size_t)N_NODES * 512, stream);
  k_edge_scatter<<<N_EDGES, 128, 0, stream>>>(src, dst, ew, dinv, xw, agg, N_EDGES, 512);
  {
    unsigned tot = (unsigned)N_NODES * 512;
    k_selfloop_bias_relu<<<(tot + 255) / 256, 256, 0, stream>>>(
        agg, xw, dinv, b1, hB, N_NODES, 9);                    // h1 (bf16) in hB
  }

  // ---- GCN layer 2: 512 -> 256 --------------------------------------------
  launch_gemm(hB, W2t, nullptr, xw, nullptr, N_NODES, 256, 512, 0, stream);
  launch_fill(agg, 0.0f, (size_t)N_NODES * 256, stream);
  k_edge_scatter<<<N_EDGES, 64, 0, stream>>>(src, dst, ew, dinv, xw, agg, N_EDGES, 256);
  {
    unsigned tot = (unsigned)N_NODES * 256;
    k_selfloop_bias_relu<<<(tot + 255) / 256, 256, 0, stream>>>(
        agg, xw, dinv, b2, hA, N_NODES, 8);                    // h2 (bf16) in hA
  }

  // ---- GCN layer 3: 256 -> 128 --------------------------------------------
  launch_gemm(hA, W3t, nullptr, xw, nullptr, N_NODES, 128, 256, 0, stream);
  launch_fill(agg, 0.0f, (size_t)N_NODES * 128, stream);
  k_edge_scatter<<<N_EDGES, 32, 0, stream>>>(src, dst, ew, dinv, xw, agg, N_EDGES, 128);
  {
    unsigned tot = (unsigned)N_NODES * 128;
    k_selfloop_bias_relu<<<(tot + 255) / 256, 256, 0, stream>>>(
        agg, xw, dinv, b3, hB, N_NODES, 7);                    // h3 (bf16) in hB
  }

  // ---- MLP head: 128 -> 64 -> 32 -> 40, then softmax -----------------------
  launch_gemm(hB, L1t, L1b, nullptr, hA, N_NODES, 64, 128, 1, stream);   // bf16+relu
  launch_gemm(hA, L2t, L2b, nullptr, hB, N_NODES, 32, 64, 1, stream);    // bf16+relu
  launch_gemm(hB, L3t, L3b, xw, nullptr, N_NODES, 40, 32, 0, stream);    // f32 logits
  k_softmax<<<(N_NODES + 255) / 256, 256, 0, stream>>>(xw, outp, N_NODES, 40);
}
